// GNN_UMLS_Model_26654567039598
// MI455X (gfx1250) — matrix operand order repaired and checked by reading, not compiled, not run on hardware
//
#include <hip/hip_runtime.h>

// ---------------------------------------------------------------------------
// MI455X (gfx1250) implementation. wave32, WMMA bf16 16x16x32 for all GEMMs.
// GEMM: 64x64 macro tile, 8 waves, 2 accumulators/wave, b128 LDS traffic.
// ---------------------------------------------------------------------------

#define HID   1024
#define NH    4
#define FF_D  2048
#define DF    768
#define SLOPE 0.2f

typedef __attribute__((ext_vector_type(16))) __bf16        v16bf;
typedef __attribute__((ext_vector_type(8)))  float         v8f;
typedef __attribute__((ext_vector_type(4)))  unsigned int  u32x4;

union FragBF { v16bf v; u32x4 q[2]; unsigned short u[16]; };

__device__ __forceinline__ unsigned short f2bf(float f) {
  unsigned u = __float_as_uint(f);
  unsigned r = u + 0x7FFFu + ((u >> 16) & 1u);   // round-to-nearest-even
  return (unsigned short)(r >> 16);
}

// Order-preserving float<->uint mapping (for atomicMax-based segment max)
__device__ __forceinline__ unsigned fkey(float f) {
  unsigned u = __float_as_uint(f);
  return (u & 0x80000000u) ? ~u : (u | 0x80000000u);
}
__device__ __forceinline__ float fkey_inv(unsigned k) {
  unsigned u = (k & 0x80000000u) ? (k & 0x7FFFFFFFu) : ~k;
  return __uint_as_float(u);
}

// ---------------------------------------------------------------------------
// f32 -> bf16 convert (grid-stride)
// ---------------------------------------------------------------------------
__global__ __launch_bounds__(256) void k_f32_to_bf16(const float* __restrict__ s,
                                                     unsigned short* __restrict__ d,
                                                     long n) {
  long i = (long)blockIdx.x * 256 + threadIdx.x;
  long stride = (long)gridDim.x * 256;
  for (; i < n; i += stride) d[i] = f2bf(s[i]);
}

// f32 [R x C] -> bf16 [C x R] transpose via padded LDS tile
__global__ __launch_bounds__(256) void k_transpose_f32_to_bf16(const float* __restrict__ s,
                                                               unsigned short* __restrict__ d,
                                                               int R, int C) {
  __shared__ float tile[32][33];
  int bx = blockIdx.x * 32, by = blockIdx.y * 32;
  int tx = threadIdx.x & 31, ty = threadIdx.x >> 5;   // 32 x 8
  for (int i = 0; i < 32; i += 8) {
    int r = by + ty + i, c = bx + tx;
    tile[ty + i][tx] = (r < R && c < C) ? s[(long)r * C + c] : 0.0f;
  }
  __syncthreads();
  for (int i = 0; i < 32; i += 8) {
    int r = bx + ty + i, c = by + tx;                 // d is C x R
    if (r < C && c < R) d[(long)r * R + c] = f2bf(tile[tx][ty + i]);
  }
}

// ---------------------------------------------------------------------------
// bf16 WMMA GEMM:  C[M,N] = act(A[M,K] @ B[K,N] + bias)
// Requirements (all call sites satisfy): K % 32 == 0, N % 8 == 0, 16B-aligned
// A and B bases. M may be ragged (row-guarded).
// block = 256 threads = 8 waves; macro tile 64(M) x 64(N); K-step 32.
// Each wave: two 16x16 accumulators (M and M+32) sharing one B fragment.
// act: 0 = none, 1 = leaky_relu(0.2), 2 = relu
// ---------------------------------------------------------------------------
__global__ __launch_bounds__(256) void k_gemm_bf16(const unsigned short* __restrict__ A,
                                                   const unsigned short* __restrict__ Bm,
                                                   const float* __restrict__ bias,
                                                   float* __restrict__ C,
                                                   int M, int N, int K, int act) {
  // row strides are multiples of 16B so b128 LDS ops stay aligned
  __shared__ __align__(16) unsigned short As[64][40];   // 64 x 32 tile, pad 8
  __shared__ __align__(16) unsigned short Bs[32][72];   // 32 x 64 tile, pad 8

  const int tid  = threadIdx.x;
  const int lane = tid & 31;
  const int wid  = tid >> 5;
  const int wrow = wid >> 2;            // 0..1
  const int wcol = wid & 3;             // 0..3
  const int tileM = blockIdx.y * 64;
  const int tileN = blockIdx.x * 64;

  // staging coordinates: one b128 per thread per operand
  const int ar = tid >> 2;              // 0..63 (A row)
  const int ac = (tid & 3) * 8;         // 0/8/16/24 (A k-chunk)
  const int br = tid >> 3;              // 0..31 (B k-row)
  const int bc = (tid & 7) * 8;         // 0..56 (B n-chunk)
  const int gmA = tileM + ar;
  const int gnB = tileN + bc;

  v8f acc0 = {0.f, 0.f, 0.f, 0.f, 0.f, 0.f, 0.f, 0.f};
  v8f acc1 = {0.f, 0.f, 0.f, 0.f, 0.f, 0.f, 0.f, 0.f};

  // fragment coordinates
  const int lr    = lane & 15;
  const int kg    = (lane >> 4) * 8;    // A: K-chunk selector per ISA 7.12.2
  const int fRow0 = wrow * 16 + lr;
  const int fRow1 = fRow0 + 32;
  const int fbCol = wcol * 16;          // B: lane%16 = N column, elements span K

  for (int k0 = 0; k0 < K; k0 += 32) {
    u32x4 av = {0u, 0u, 0u, 0u};
    if (gmA < M) av = *(const u32x4*)&A[(long)gmA * K + (k0 + ac)];
    u32x4 bv = {0u, 0u, 0u, 0u};
    if (gnB < N) bv = *(const u32x4*)&Bm[(long)(k0 + br) * N + gnB];

    if (k0 + 32 < K) {                   // prefetch next K tile into caches
      if (gmA < M) __builtin_prefetch(&A[(long)gmA * K + (k0 + 32 + ac)], 0, 0);
      if (gnB < N) __builtin_prefetch(&Bm[(long)(k0 + 32 + br) * N + gnB], 0, 0);
    }

    __syncthreads();                     // previous iteration readers done
    *(u32x4*)&As[ar][ac] = av;
    *(u32x4*)&Bs[br][bc] = bv;
    __syncthreads();

    FragBF fa0, fa1, fb;
    fa0.q[0] = *(const u32x4*)&As[fRow0][kg];
    fa0.q[1] = *(const u32x4*)&As[fRow0][16 + kg];
    fa1.q[0] = *(const u32x4*)&As[fRow1][kg];
    fa1.q[1] = *(const u32x4*)&As[fRow1][16 + kg];
    for (int s = 0; s < 8; ++s) {
      fb.u[s]     = Bs[kg + s][fbCol + lr];
      fb.u[8 + s] = Bs[16 + kg + s][fbCol + lr];
    }

    acc0 = __builtin_amdgcn_wmma_f32_16x16x32_bf16(false, fa0.v, false, fb.v,
                                                   (short)0, acc0, false, false);
    acc1 = __builtin_amdgcn_wmma_f32_16x16x32_bf16(false, fa1.v, false, fb.v,
                                                   (short)0, acc1, false, false);
  }

  // C/D layout: lane -> N=lane%16, VGPR r -> M = r + 8*(lane/16)
  const int col = tileN + wcol * 16 + (lane & 15);
  const int rb  = tileM + wrow * 16 + (lane >> 4) * 8;
  const float bval = (bias != nullptr && col < N) ? bias[col] : 0.0f;
#pragma unroll
  for (int r = 0; r < 8; ++r) {
    int row = rb + r;
    if (row < M && col < N) {
      float x = acc0[r] + bval;
      if (act == 1)      x = x > 0.f ? x : SLOPE * x;
      else if (act == 2) x = x > 0.f ? x : 0.f;
      C[(long)row * N + col] = x;
    }
  }
#pragma unroll
  for (int r = 0; r < 8; ++r) {
    int row = rb + 32 + r;
    if (row < M && col < N) {
      float x = acc1[r] + bval;
      if (act == 1)      x = x > 0.f ? x : SLOPE * x;
      else if (act == 2) x = x > 0.f ? x : 0.f;
      C[(long)row * N + col] = x;
    }
  }
}

// ---------------------------------------------------------------------------
// GATv2 edge kernels. edge_index: [0:E]=dst, [E:2E]=src
// ---------------------------------------------------------------------------
__global__ __launch_bounds__(256) void k_gat_alpha(const float* __restrict__ xl,
                                                   const float* __restrict__ xr,
                                                   const int* __restrict__ ei,
                                                   const float* __restrict__ att,
                                                   unsigned* __restrict__ amax,
                                                   float* __restrict__ aexp,
                                                   int E, int nh, int hd) {
  int idx = blockIdx.x * 256 + threadIdx.x;
  if (idx >= E * nh) return;
  int e = idx / nh, h = idx - e * nh;
  int dst = ei[e], src = ei[E + e];
  const float* a = att + h * hd;
  const float* l = xl + (long)src * nh * hd + h * hd;
  const float* r = xr + (long)dst * nh * hd + h * hd;
  float s = 0.f;
  for (int c = 0; c < hd; ++c) {
    float v = l[c] + r[c];
    v = v > 0.f ? v : SLOPE * v;           // leaky_relu before att dot
    s += v * a[c];
  }
  aexp[idx] = s;
  atomicMax(&amax[dst * nh + h], fkey(s));
}

__global__ __launch_bounds__(256) void k_gat_exp(float* __restrict__ aexp,
                                                 const int* __restrict__ ei,
                                                 const unsigned* __restrict__ amax,
                                                 float* __restrict__ denom,
                                                 int E, int nh) {
  int idx = blockIdx.x * 256 + threadIdx.x;
  if (idx >= E * nh) return;
  int e = idx / nh, h = idx - e * nh;
  int dst = ei[e];
  float a = __expf(aexp[idx] - fkey_inv(amax[dst * nh + h]));
  aexp[idx] = a;
  atomicAdd(&denom[dst * nh + h], a);
}

__global__ __launch_bounds__(256) void k_gat_aggr(const float* __restrict__ xl,
                                                  const int* __restrict__ ei,
                                                  const float* __restrict__ aexp,
                                                  const float* __restrict__ denom,
                                                  float* __restrict__ aggr,
                                                  int E, int nh, int hd) {
  long idx = (long)blockIdx.x * 256 + threadIdx.x;
  long total = (long)E * nh * hd;
  if (idx >= total) return;
  int c = (int)(idx % hd);
  long t2 = idx / hd;
  int h = (int)(t2 % nh);
  int e = (int)(t2 / nh);
  int dst = ei[e], src = ei[E + e];
  float w = aexp[(long)e * nh + h] / (denom[dst * nh + h] + 1e-16f);
  atomicAdd(&aggr[(long)dst * nh * hd + h * hd + c],
            xl[(long)src * nh * hd + h * hd + c] * w);
}

__global__ __launch_bounds__(256) void k_bias_act(float* __restrict__ x,
                                                  const float* __restrict__ bias,
                                                  long total, int N, int act) {
  long i = (long)blockIdx.x * 256 + threadIdx.x;
  if (i >= total) return;
  float v = x[i] + bias[i % N];
  if (act == 1) v = v > 0.f ? v : SLOPE * v;
  x[i] = v;
}

// ---------------------------------------------------------------------------
// Small glue kernels
// ---------------------------------------------------------------------------
__global__ __launch_bounds__(256) void k_gather(const float* __restrict__ x,
                                                const int* __restrict__ idxv,
                                                float* __restrict__ t, int L, int N) {
  int i = blockIdx.x * 256 + threadIdx.x;
  if (i >= L * N) return;
  int l = i / N, c = i - l * N;
  t[i] = x[(long)idxv[l] * N + c];
}

// resid[b,l,:] = t[l,:] + y[b,:]   (cross-attn collapses: softmax over 1 key == 1)
__global__ __launch_bounds__(256) void k_xresid(const float* __restrict__ t,
                                                const float* __restrict__ y,
                                                float* __restrict__ o,
                                                int B, int L, int H) {
  long i = (long)blockIdx.x * 256 + threadIdx.x;
  long tot = (long)B * L * H;
  if (i >= tot) return;
  int c = (int)(i % H);
  long r = i / H;
  int l = (int)(r % L);
  int b = (int)(r / L);
  o[i] = t[(long)l * H + c] + y[(long)b * H + c];
}

// o = LayerNorm(x + y), one block per row; y may be nullptr; o may alias x
__global__ __launch_bounds__(256) void k_add_ln(const float* __restrict__ x,
                                                const float* __restrict__ y,
                                                const float* __restrict__ g,
                                                const float* __restrict__ b,
                                                float* __restrict__ o, int H) {
  __shared__ float red[256];
  long row = blockIdx.x;
  const float* xr = x + row * (long)H;
  const float* yr = y ? y + row * (long)H : nullptr;

  float s = 0.f;
  for (int c = threadIdx.x; c < H; c += 256) s += xr[c] + (yr ? yr[c] : 0.f);
  red[threadIdx.x] = s; __syncthreads();
  for (int off = 128; off > 0; off >>= 1) {
    if ((int)threadIdx.x < off) red[threadIdx.x] += red[threadIdx.x + off];
    __syncthreads();
  }
  float mean = red[0] / H; __syncthreads();

  float vs = 0.f;
  for (int c = threadIdx.x; c < H; c += 256) {
    float v = xr[c] + (yr ? yr[c] : 0.f) - mean;
    vs += v * v;
  }
  red[threadIdx.x] = vs; __syncthreads();
  for (int off = 128; off > 0; off >>= 1) {
    if ((int)threadIdx.x < off) red[threadIdx.x] += red[threadIdx.x + off];
    __syncthreads();
  }
  float inv = rsqrtf(red[0] / H + 1e-5f); __syncthreads();

  for (int c = threadIdx.x; c < H; c += 256) {
    float v = xr[c] + (yr ? yr[c] : 0.f);
    o[row * (long)H + c] = (v - mean) * inv * g[c] + b[c];
  }
}

// encoder self-attention: seq len <= 16, one block per (batch, head)
__global__ __launch_bounds__(256) void k_mha(const float* __restrict__ qkv,
                                             float* __restrict__ o,
                                             int B, int L, int nh, int hd) {
  __shared__ float sS[16][16];
  __shared__ float sP[16][16];
  int b = blockIdx.x / nh, h = blockIdx.x - b * nh;
  int H = nh * hd, H3 = 3 * H;
  const float* base = qkv + (long)b * L * H3;
  int tid = threadIdx.x;

  for (int p = tid; p < L * L; p += 256) {
    int i = p / L, j = p - i * L;
    const float* q = base + (long)i * H3 + h * hd;
    const float* k = base + (long)j * H3 + H + h * hd;
    float s = 0.f;
    for (int d = 0; d < hd; ++d) s += q[d] * k[d];
    sS[i][j] = s * rsqrtf((float)hd);
  }
  __syncthreads();
  if (tid < L) {
    float mx = -3.4e38f;
    for (int j = 0; j < L; ++j) mx = fmaxf(mx, sS[tid][j]);
    float sum = 0.f;
    for (int j = 0; j < L; ++j) { float e = __expf(sS[tid][j] - mx); sP[tid][j] = e; sum += e; }
    float inv = 1.f / sum;
    for (int j = 0; j < L; ++j) sP[tid][j] *= inv;
  }
  __syncthreads();
  for (int p = tid; p < L * hd; p += 256) {
    int i = p / hd, d = p - i * hd;
    float s = 0.f;
    for (int j = 0; j < L; ++j) s += sP[i][j] * base[(long)j * H3 + 2 * H + h * hd + d];
    o[((long)(b * L + i)) * H + h * hd + d] = s;
  }
}

__global__ __launch_bounds__(256) void k_cls(const float* __restrict__ src,
                                             const float* __restrict__ w,
                                             const float* __restrict__ b,
                                             float* __restrict__ out, int H) {
  __shared__ float red[256];
  long row = blockIdx.x;
  float s = 0.f;
  for (int c = threadIdx.x; c < H; c += 256) s += src[row * (long)H + c] * w[c];
  red[threadIdx.x] = s; __syncthreads();
  for (int off = 128; off > 0; off >>= 1) {
    if ((int)threadIdx.x < off) red[threadIdx.x] += red[threadIdx.x + off];
    __syncthreads();
  }
  if (threadIdx.x == 0) out[row] = red[0] + b[0];
}

// ---------------------------------------------------------------------------
// Host-side helpers
// ---------------------------------------------------------------------------
static inline void cvt(hipStream_t st, const float* s, unsigned short* d, long n) {
  long blocks = (n + 255) / 256;
  if (blocks > 1048576) blocks = 1048576;       // grid-stride handles the rest
  k_f32_to_bf16<<<dim3((unsigned)blocks), 256, 0, st>>>(s, d, n);
}
static inline void tcvt(hipStream_t st, const float* s, unsigned short* d, int R, int C) {
  dim3 g((C + 31) / 32, (R + 31) / 32);
  k_transpose_f32_to_bf16<<<g, 256, 0, st>>>(s, d, R, C);
}
static inline void gemm(hipStream_t st, const unsigned short* A, const unsigned short* Bw,
                        const float* bias, float* C, int M, int N, int K, int act) {
  dim3 g((N + 63) / 64, (M + 63) / 64);
  k_gemm_bf16<<<g, 256, 0, st>>>(A, Bw, bias, C, M, N, K, act);
}

extern "C" void kernel_launch(void* const* d_in, const int* in_sizes, int n_in,
                              void* d_out, int out_size, void* d_ws, size_t ws_size,
                              hipStream_t stream) {
  // ---- inputs (setup_inputs order) ----
  const float* img      = (const float*)d_in[0];
  const float* nf       = (const float*)d_in[1];
  const int*   ei       = (const int*)d_in[2];
  const int*   tidx     = (const int*)d_in[3];
  const float* g1wl     = (const float*)d_in[4];
  const float* g1bl     = (const float*)d_in[5];
  const float* g1wr     = (const float*)d_in[6];
  const float* g1br     = (const float*)d_in[7];
  const float* g1att    = (const float*)d_in[8];
  const float* g1bias   = (const float*)d_in[9];
  const float* g2wl     = (const float*)d_in[10];
  const float* g2bl     = (const float*)d_in[11];
  const float* g2wr     = (const float*)d_in[12];
  const float* g2br     = (const float*)d_in[13];
  const float* g2att    = (const float*)d_in[14];
  const float* g2bias   = (const float*)d_in[15];
  const float* xa_in_w  = (const float*)d_in[16];
  const float* xa_in_b  = (const float*)d_in[17];
  const float* xa_out_w = (const float*)d_in[18];
  const float* xa_out_b = (const float*)d_in[19];
  const float* xl_l1_w  = (const float*)d_in[20];
  const float* xl_l1_b  = (const float*)d_in[21];
  const float* xl_l2_w  = (const float*)d_in[22];
  const float* xl_l2_b  = (const float*)d_in[23];
  const float* xl_n1_g  = (const float*)d_in[24];
  const float* xl_n1_b  = (const float*)d_in[25];
  const float* xl_n2_g  = (const float*)d_in[26];
  const float* xl_n2_b  = (const float*)d_in[27];
  const float* enc_in_w = (const float*)d_in[28];
  const float* enc_in_b = (const float*)d_in[29];
  const float* enc_out_w= (const float*)d_in[30];
  const float* enc_out_b= (const float*)d_in[31];
  const float* enc_l1_w = (const float*)d_in[32];
  const float* enc_l1_b = (const float*)d_in[33];
  const float* enc_l2_w = (const float*)d_in[34];
  const float* enc_l2_b = (const float*)d_in[35];
  const float* enc_n1_g = (const float*)d_in[36];
  const float* enc_n1_b = (const float*)d_in[37];
  const float* enc_n2_g = (const float*)d_in[38];
  const float* enc_n2_b = (const float*)d_in[39];
  const float* cls_w    = (const float*)d_in[40];
  const float* cls_b    = (const float*)d_in[41];

  const int B  = in_sizes[0] / HID;     // 32
  const int Nn = in_sizes[1] / DF;      // 8000
  const int E  = in_sizes[2] / 2;       // 104000
  const int L  = in_sizes[3];           // 14
  const int BL = B * L;                 // 448
  const int H3 = 3 * HID;
  const int hd1 = HID / NH;             // 256

  // ---- workspace carve (bump allocator) ----
  size_t off = 0;
  char* ws = (char*)d_ws;
  auto carve = [&](size_t bytes) -> char* {
    char* p = ws + off;
    off += (bytes + 255) & ~(size_t)255;
    return p;
  };
  unsigned short* abf   = (unsigned short*)carve((size_t)Nn * HID * 2);  // activations bf16
  unsigned short* wbf   = (unsigned short*)carve((size_t)H3 * HID * 2);  // weights bf16
  float* xl_b  = (float*)carve((size_t)Nn * HID * 4);
  float* xr_b  = (float*)carve((size_t)Nn * HID * 4);
  float* aggr1 = (float*)carve((size_t)Nn * HID * 4);
  float* aggr2 = (float*)carve((size_t)Nn * HID * 4);
  float* aexp  = (float*)carve((size_t)E * NH * 4);
  unsigned* amax = (unsigned*)carve((size_t)Nn * NH * 4);
  float* denom = (float*)carve((size_t)Nn * NH * 4);
  float* tbuf  = (float*)carve((size_t)L * HID * 4);
  float* vh    = (float*)carve((size_t)B * HID * 4);
  float* yv    = (float*)carve((size_t)B * HID * 4);
  float* srcb  = (float*)carve((size_t)BL * HID * 4);
  float* resid = (float*)carve((size_t)BL * HID * 4);
  float* ff1   = (float*)carve((size_t)BL * FF_D * 4);
  float* ff2   = (float*)carve((size_t)BL * HID * 4);
  float* qkv   = (float*)carve((size_t)BL * H3 * 4);
  float* attn  = (float*)carve((size_t)BL * HID * 4);
  (void)ws_size; (void)n_in; (void)out_size;

  // ================= GAT layer 1 (NH=4, hd=256) =================
  cvt(stream, nf, abf, (long)Nn * DF);
  cvt(stream, g1wl, wbf, (long)DF * HID);
  gemm(stream, abf, wbf, g1bl, xl_b, Nn, HID, DF, 0);
  cvt(stream, g1wr, wbf, (long)DF * HID);
  gemm(stream, abf, wbf, g1br, xr_b, Nn, HID, DF, 0);

  hipMemsetAsync(amax, 0, (size_t)Nn * NH * 4, stream);     // fkey order: 0 == -max
  hipMemsetAsync(denom, 0, (size_t)Nn * NH * 4, stream);
  hipMemsetAsync(aggr1, 0, (size_t)Nn * HID * 4, stream);

  {
    int total = E * NH;
    k_gat_alpha<<<(total + 255) / 256, 256, 0, stream>>>(xl_b, xr_b, ei, g1att, amax, aexp, E, NH, hd1);
    k_gat_exp  <<<(total + 255) / 256, 256, 0, stream>>>(aexp, ei, amax, denom, E, NH);
    long tot3 = (long)E * NH * hd1;
    k_gat_aggr <<<(unsigned)((tot3 + 255) / 256), 256, 0, stream>>>(xl_b, ei, aexp, denom, aggr1, E, NH, hd1);
    long nt = (long)Nn * HID;
    k_bias_act <<<(unsigned)((nt + 255) / 256), 256, 0, stream>>>(aggr1, g1bias, nt, HID, 1);
  }

  // ================= GAT layer 2 (heads=1, hd=1024) =================
  cvt(stream, aggr1, abf, (long)Nn * HID);
  cvt(stream, g2wl, wbf, (long)HID * HID);
  gemm(stream, abf, wbf, g2bl, xl_b, Nn, HID, HID, 0);
  cvt(stream, g2wr, wbf, (long)HID * HID);
  gemm(stream, abf, wbf, g2br, xr_b, Nn, HID, HID, 0);

  hipMemsetAsync(amax, 0, (size_t)Nn * 1 * 4, stream);
  hipMemsetAsync(denom, 0, (size_t)Nn * 1 * 4, stream);
  hipMemsetAsync(aggr2, 0, (size_t)Nn * HID * 4, stream);

  {
    int total = E;
    k_gat_alpha<<<(total + 255) / 256, 256, 0, stream>>>(xl_b, xr_b, ei, g2att, amax, aexp, E, 1, HID);
    k_gat_exp  <<<(total + 255) / 256, 256, 0, stream>>>(aexp, ei, amax, denom, E, 1);
    long tot3 = (long)E * HID;
    k_gat_aggr <<<(unsigned)((tot3 + 255) / 256), 256, 0, stream>>>(xl_b, ei, aexp, denom, aggr2, E, 1, HID);
    long nt = (long)Nn * HID;
    k_bias_act <<<(unsigned)((nt + 255) / 256), 256, 0, stream>>>(aggr2, g2bias, nt, HID, 0);
  }

  // t = x[target_idx]
  k_gather<<<(L * HID + 255) / 256, 256, 0, stream>>>(aggr2, tidx, tbuf, L, HID);

  // ============== cross-attention (kv len == 1 -> softmax == 1) ==============
  // attn[b,l,:] = out_proj(v_proj(img[b])) ; independent of q.
  cvt(stream, img, abf, (long)B * HID);
  tcvt(stream, xa_in_w + (size_t)2 * HID * HID, wbf, HID, HID);      // Wv rows -> K x N
  gemm(stream, abf, wbf, xa_in_b + 2 * HID, vh, B, HID, HID, 0);
  cvt(stream, vh, abf, (long)B * HID);
  tcvt(stream, xa_out_w, wbf, HID, HID);
  gemm(stream, abf, wbf, xa_out_b, yv, B, HID, HID, 0);

  {
    long tot = (long)BL * HID;
    k_xresid<<<(unsigned)((tot + 255) / 256), 256, 0, stream>>>(tbuf, yv, resid, B, L, HID);
  }
  k_add_ln<<<BL, 256, 0, stream>>>(resid, nullptr, xl_n1_g, xl_n1_b, srcb, HID);

  // ================= xl feed-forward =================
  cvt(stream, srcb, abf, (long)BL * HID);
  tcvt(stream, xl_l1_w, wbf, FF_D, HID);
  gemm(stream, abf, wbf, xl_l1_b, ff1, BL, FF_D, HID, 2);
  cvt(stream, ff1, abf, (long)BL * FF_D);
  tcvt(stream, xl_l2_w, wbf, HID, FF_D);
  gemm(stream, abf, wbf, xl_l2_b, ff2, BL, HID, FF_D, 0);
  k_add_ln<<<BL, 256, 0, stream>>>(srcb, ff2, xl_n2_g, xl_n2_b, srcb, HID);

  // ================= 2 encoder layers =================
  for (int i = 0; i < 2; ++i) {
    cvt(stream, srcb, abf, (long)BL * HID);
    tcvt(stream, enc_in_w + (size_t)i * H3 * HID, wbf, H3, HID);
    gemm(stream, abf, wbf, enc_in_b + (size_t)i * H3, qkv, BL, H3, HID, 0);

    k_mha<<<B * NH, 256, 0, stream>>>(qkv, attn, B, L, NH, hd1);

    cvt(stream, attn, abf, (long)BL * HID);
    tcvt(stream, enc_out_w + (size_t)i * HID * HID, wbf, HID, HID);
    gemm(stream, abf, wbf, enc_out_b + (size_t)i * HID, ff2, BL, HID, HID, 0);
    k_add_ln<<<BL, 256, 0, stream>>>(srcb, ff2, enc_n1_g + (size_t)i * HID,
                                     enc_n1_b + (size_t)i * HID, srcb, HID);

    cvt(stream, srcb, abf, (long)BL * HID);
    tcvt(stream, enc_l1_w + (size_t)i * FF_D * HID, wbf, FF_D, HID);
    gemm(stream, abf, wbf, enc_l1_b + (size_t)i * FF_D, ff1, BL, FF_D, HID, 2);
    cvt(stream, ff1, abf, (long)BL * FF_D);
    tcvt(stream, enc_l2_w + (size_t)i * HID * FF_D, wbf, HID, FF_D);
    gemm(stream, abf, wbf, enc_l2_b + (size_t)i * HID, ff2, BL, HID, FF_D, 0);
    k_add_ln<<<BL, 256, 0, stream>>>(srcb, ff2, enc_n2_g + (size_t)i * HID,
                                     enc_n2_b + (size_t)i * HID, srcb, HID);
  }

  // ================= classifier =================
  k_cls<<<BL, 256, 0, stream>>>(srcb, cls_w, cls_b, (float*)d_out, HID);
}
